// Attn_16527034155411
// MI455X (gfx1250) — matrix operand order repaired and checked by reading, not compiled, not run on hardware
//
#include <hip/hip_runtime.h>

// ---------------------------------------------------------------------------
// Types for CDNA5 WMMA (wave32)
// ---------------------------------------------------------------------------
typedef __attribute__((ext_vector_type(16))) __bf16 v16bf;
typedef __attribute__((ext_vector_type(8)))  float  v8f;
typedef __attribute__((ext_vector_type(4)))  unsigned int u32x4;
typedef __attribute__((ext_vector_type(8)))  int i32x8;
typedef __attribute__((ext_vector_type(4)))  int i32x4;

union FragBF {
    v16bf v;
    unsigned short u[16];
    unsigned int   w[8];
};
union AccF {
    v8f   v;
    float f[8];
};

__device__ __forceinline__ unsigned short f2bf(float f) {
    union { float f; unsigned int u; } x;
    x.f = f;
    unsigned int r = x.u + 0x7FFFu + ((x.u >> 16) & 1u);  // round-to-nearest-even
    return (unsigned short)(r >> 16);
}

// ---------------------------------------------------------------------------
// Tensor Data Mover: 2D bf16 tile  global -> LDS  (CDNA5 TDM, ISA ch.8)
// ---------------------------------------------------------------------------
#if __has_builtin(__builtin_amdgcn_tensor_load_to_lds) && \
    __has_builtin(__builtin_amdgcn_s_wait_tensorcnt)
#define USE_TDM 1
#else
#define USE_TDM 0
#endif

#if USE_TDM
// tile_d0: contiguous elements per row; tile_d1: rows; stride_d0: tensor row
// stride (elements); pad_interval/pad_amount per D# encoding (LDS row padding).
__device__ __forceinline__ void tdm_load_2d_bf16(
    unsigned int lds_addr, const void* gptr,
    unsigned int tile_d0, unsigned int tile_d1,
    unsigned int tensor_d0, unsigned int tensor_d1,
    unsigned int stride_d0, unsigned int pad_interval, unsigned int pad_amount) {
    unsigned long long ga = (unsigned long long)(uintptr_t)gptr;
    u32x4 g0;
    g0[0] = 1u;                                     // count=1, user descriptor
    g0[1] = lds_addr;                               // lds_addr [63:32]
    g0[2] = (unsigned int)(ga & 0xFFFFFFFFull);     // global_addr [95:64]
    g0[3] = (unsigned int)((ga >> 32) & 0x1FFFFFFull) | (2u << 30);  // type=2
    i32x8 g1;
    g1[0] = (int)((1u << 16)                        // data_size = 2 bytes
                  | (1u << 20)                      // pad_enable
                  | (pad_interval << 22)
                  | (pad_amount << 25));
    g1[1] = (int)((tensor_d0 & 0xFFFFu) << 16);                 // dim0 lo
    g1[2] = (int)((tensor_d0 >> 16) | ((tensor_d1 & 0xFFFFu) << 16));
    g1[3] = (int)((tensor_d1 >> 16) | (tile_d0 << 16));
    g1[4] = (int)(tile_d1 & 0xFFFFu);               // tile_dim1 (tile_dim2=0)
    g1[5] = (int)stride_d0;                         // dim0 stride lo
    g1[6] = 0;                                      // stride hi / dim1 stride
    g1[7] = 0;
    i32x4 z4 = {0, 0, 0, 0};
#if __clang_major__ >= 23
    i32x8 z8 = {0, 0, 0, 0, 0, 0, 0, 0};
    __builtin_amdgcn_tensor_load_to_lds(g0, g1, z4, z4, z8, 0);
#else
    __builtin_amdgcn_tensor_load_to_lds(g0, g1, z4, z4, 0);
#endif
}
#endif

// ---------------------------------------------------------------------------
// f32 -> bf16 conversion
// ---------------------------------------------------------------------------
__global__ void cvt_f32_bf16(const float* __restrict__ src,
                             unsigned short* __restrict__ dst, int n) {
    int i = blockIdx.x * 256 + threadIdx.x;
    if (i < n) dst[i] = f2bf(src[i]);
}

// ---------------------------------------------------------------------------
// Tiled bf16 GEMM with WMMA.  C[M,N] = A[M,K] * B[K,N] + bias.
// Block: 256 threads = 8 waves; block tile 128x64; wave tile 32x32; K-step 32.
// Tiles staged via TDM (tensor_load_to_lds) when available.
// mode 0: QKV epilogue (scatter to Q/K/V head layouts + present f32)
// mode 1: plain f32 row-major output (final projection)
// ---------------------------------------------------------------------------
__global__ __launch_bounds__(256) void gemm_bf16_kernel(
    const unsigned short* __restrict__ A, const unsigned short* __restrict__ Bm,
    const float* __restrict__ bias, int M, int N, int K, int mode,
    unsigned short* __restrict__ q_bf, unsigned short* __restrict__ k_bf,
    unsigned short* __restrict__ vT_bf, float* __restrict__ present,
    float* __restrict__ outA) {
    __shared__ __align__(16) unsigned short As[128 * 40];  // 128x32, row pad 8
    __shared__ __align__(16) unsigned short Bs[32 * 72];   // 32x64,  row pad 8

    const int tid  = threadIdx.x;
    const int lane = tid & 31;
    const int wv   = tid >> 5;
    const int half = lane >> 4;
    const int l16  = lane & 15;
    const int m_blk = blockIdx.x * 128;
    const int n_blk = blockIdx.y * 64;
    const int wm = (wv & 3) * 32;   // wave M offset inside block tile
    const int wn = (wv >> 2) * 32;  // wave N offset inside block tile

    AccF acc[2][2];
#pragma unroll
    for (int mi = 0; mi < 2; ++mi)
#pragma unroll
        for (int ni = 0; ni < 2; ++ni)
#pragma unroll
            for (int r = 0; r < 8; ++r) acc[mi][ni].f[r] = 0.0f;

    for (int k0 = 0; k0 < K; k0 += 32) {
#if USE_TDM
        // ---- TDM staging: one descriptor per tile, issued by wave 0 ----
        if (wv == 0) {
            // A tile: 128 rows x 32 cols, LDS rows 64B + 16B pad
            tdm_load_2d_bf16((unsigned int)(uintptr_t)(void*)&As[0],
                             A + (size_t)m_blk * K + k0,
                             /*tile*/ 32, 128, /*tensor*/ (unsigned)K, 128,
                             /*stride*/ (unsigned)K,
                             /*pad_interval(16 dw)*/ 3, /*pad(4 dw)*/ 3);
            // B tile: 32 rows x 64 cols, LDS rows 128B + 16B pad
            tdm_load_2d_bf16((unsigned int)(uintptr_t)(void*)&Bs[0],
                             Bm + (size_t)k0 * N + n_blk,
                             /*tile*/ 64, 32, /*tensor*/ (unsigned)N, 32,
                             /*stride*/ (unsigned)N,
                             /*pad_interval(32 dw)*/ 4, /*pad(4 dw)*/ 3);
            __builtin_amdgcn_s_wait_tensorcnt(0);
        }
        __syncthreads();
#else
        // ---- cooperative LDS staging (vectorized 16B) ----
#pragma unroll
        for (int i = 0; i < 2; ++i) {
            int e   = tid + i * 256;       // 512 chunks of 8 bf16 for A tile
            int row = e >> 2;
            int ch  = e & 3;
            *(uint4*)&As[row * 40 + ch * 8] =
                *(const uint4*)(A + (size_t)(m_blk + row) * K + k0 + ch * 8);
        }
        {
            int row = tid >> 3;            // 256 chunks of 8 bf16 for B tile
            int ch  = tid & 7;
            *(uint4*)&Bs[row * 72 + ch * 8] =
                *(const uint4*)(Bm + (size_t)(k0 + row) * N + n_blk + ch * 8);
        }
        __syncthreads();
#endif

        // ---- fragments (contiguous k = 16*half + s in A and B) ----
        FragBF af[2], bfragm[2];
#pragma unroll
        for (int mi = 0; mi < 2; ++mi) {
            int row = wm + mi * 16 + l16;
#pragma unroll
            for (int v = 0; v < 8; ++v) {
                int kk = (half << 4) + (v << 1);
                af[mi].w[v] = *(const unsigned int*)&As[row * 40 + kk];
            }
        }
#pragma unroll
        for (int ni = 0; ni < 2; ++ni) {
            int col = wn + ni * 16 + l16;
#pragma unroll
            for (int e = 0; e < 16; ++e) {
                int kk = (half << 4) + e;
                bfragm[ni].u[e] = Bs[kk * 72 + col];
            }
        }
#pragma unroll
        for (int mi = 0; mi < 2; ++mi)
#pragma unroll
            for (int ni = 0; ni < 2; ++ni)
                acc[mi][ni].v = __builtin_amdgcn_wmma_f32_16x16x32_bf16(
                    false, af[mi].v, false, bfragm[ni].v, (short)0,
                    acc[mi][ni].v, false, false);
        __syncthreads();
    }

    // ---- epilogue ----
#pragma unroll
    for (int mi = 0; mi < 2; ++mi)
#pragma unroll
        for (int ni = 0; ni < 2; ++ni)
#pragma unroll
            for (int r = 0; r < 8; ++r) {
                int gm = m_blk + wm + mi * 16 + r + half * 8;
                int gn = n_blk + wn + ni * 16 + l16;
                float val = acc[mi][ni].f[r] + bias[gn];
                if (mode == 0) {
                    int part = gn >> 10;     // 0=q 1=k 2=v
                    int dd = gn & 1023;
                    int hh = dd >> 6;
                    int cc = dd & 63;
                    int bb = gm >> 11;
                    int ss = gm & 2047;
                    size_t hidx = ((size_t)(bb * 16 + hh) * 2048 + ss) * 64 + cc;
                    if (part == 0) {
                        q_bf[hidx] = f2bf(val);
                    } else if (part == 1) {
                        present[((size_t)((bb * 2 + 0) * 16 + hh) * 2048 + ss) * 64 + cc] = val;
                        k_bf[hidx] = f2bf(val);
                    } else {
                        present[((size_t)((bb * 2 + 1) * 16 + hh) * 2048 + ss) * 64 + cc] = val;
                        vT_bf[((size_t)(bb * 16 + hh) * 64 + cc) * 2048 + ss] = f2bf(val);
                    }
                } else {
                    outA[(size_t)gm * N + gn] = val;
                }
            }
}

// ---------------------------------------------------------------------------
// Flash attention, causal.  One wave per 16-row query tile; KV step 32.
// Q,K layout: [B,H,S,64] bf16.  V layout: [B,H,64,S] bf16 (transposed).
// Writes merged-head bf16 activations [B*S, 1024] for the final projection.
// ---------------------------------------------------------------------------
__global__ __launch_bounds__(256) void attn_kernel(
    const unsigned short* __restrict__ Q, const unsigned short* __restrict__ Kh,
    const unsigned short* __restrict__ VT, unsigned short* __restrict__ Aout) {
    __shared__ __align__(16) unsigned short Pl[8][16 * 32];  // per-wave P tile

    const int tid  = threadIdx.x;
    const int lane = tid & 31;
    const int wv   = tid >> 5;
    const int half = lane >> 4;
    const int l16  = lane & 15;

    int w  = blockIdx.x * 8 + wv;  // 0..4095 = B*H*(S/16)
    int qt = w & 127;
    int h  = (w >> 7) & 15;
    int b  = w >> 11;
    int q0 = qt * 16;

    const unsigned short* qp = Q + (size_t)(b * 16 + h) * 2048 * 64;
    const unsigned short* kp = Kh + (size_t)(b * 16 + h) * 2048 * 64;
    const unsigned short* vp = VT + (size_t)(b * 16 + h) * 64 * 2048;

    // Q A-fragments for d-halves 0..31 and 32..63 (constant over kv loop)
    FragBF aq[2];
#pragma unroll
    for (int kk = 0; kk < 2; ++kk)
#pragma unroll
        for (int v = 0; v < 8; ++v) {
            int kd = kk * 32 + (half << 4) + (v << 1);
            aq[kk].w[v] = *(const unsigned int*)(qp + (size_t)(q0 + l16) * 64 + kd);
        }

    AccF O[4];
#pragma unroll
    for (int t = 0; t < 4; ++t)
#pragma unroll
        for (int r = 0; r < 8; ++r) O[t].f[r] = 0.0f;

    float mrow[8], lrow[8];
#pragma unroll
    for (int r = 0; r < 8; ++r) { mrow[r] = -1e30f; lrow[r] = 0.0f; }

    const float scale = 0.125f;  // rsqrt(64)

    for (int kv0 = 0; kv0 < q0 + 16; kv0 += 32) {
        // ---- S = Q * K^T  (two 16x16 score tiles) ----
        AccF S[2];
#pragma unroll
        for (int j = 0; j < 2; ++j) {
#pragma unroll
            for (int r = 0; r < 8; ++r) S[j].f[r] = 0.0f;
            int n = kv0 + j * 16 + l16;
#pragma unroll
            for (int dh = 0; dh < 2; ++dh) {
                FragBF bk;
#pragma unroll
                for (int v = 0; v < 8; ++v) {
                    int kd = dh * 32 + (half << 4) + (v << 1);
                    bk.w[v] = *(const unsigned int*)(kp + (size_t)n * 64 + kd);
                }
                S[j].v = __builtin_amdgcn_wmma_f32_16x16x32_bf16(
                    false, aq[dh].v, false, bk.v, (short)0, S[j].v, false, false);
            }
        }

        // ---- scale, causal mask, online softmax ----
        float mnew[8], alpha[8];
#pragma unroll
        for (int r = 0; r < 8; ++r) {
            int row = q0 + r + half * 8;
#pragma unroll
            for (int j = 0; j < 2; ++j) {
                int col = kv0 + j * 16 + l16;
                float s = S[j].f[r] * scale;
                if (col > row) s = -1e30f;
                S[j].f[r] = s;
            }
            float t = fmaxf(S[0].f[r], S[1].f[r]);
            t = fmaxf(t, __shfl_xor(t, 1));
            t = fmaxf(t, __shfl_xor(t, 2));
            t = fmaxf(t, __shfl_xor(t, 4));
            t = fmaxf(t, __shfl_xor(t, 8));
            mnew[r]  = fmaxf(mrow[r], t);
            alpha[r] = __expf(mrow[r] - mnew[r]);
            mrow[r]  = mnew[r];
        }
#pragma unroll
        for (int r = 0; r < 8; ++r) {
            float p0 = __expf(S[0].f[r] - mnew[r]);
            float p1 = __expf(S[1].f[r] - mnew[r]);
            Pl[wv][(r + half * 8) * 32 + l16]      = f2bf(p0);
            Pl[wv][(r + half * 8) * 32 + 16 + l16] = f2bf(p1);
            float rs = p0 + p1;
            rs += __shfl_xor(rs, 1);
            rs += __shfl_xor(rs, 2);
            rs += __shfl_xor(rs, 4);
            rs += __shfl_xor(rs, 8);
            lrow[r] = lrow[r] * alpha[r] + rs;
#pragma unroll
            for (int t = 0; t < 4; ++t) O[t].f[r] *= alpha[r];
        }

        // same-wave LDS round-trip: hardware keeps DS ops in order within a
        // wave; just stop the compiler reordering across the exchange.
        asm volatile("" ::: "memory");
#if __has_builtin(__builtin_amdgcn_s_wait_dscnt)
        __builtin_amdgcn_s_wait_dscnt(0);
#endif

        // ---- P A-fragment from LDS (C-layout -> A-layout transpose) ----
        FragBF ap;
#pragma unroll
        for (int v = 0; v < 8; ++v) {
            int kk = (half << 4) + (v << 1);
            ap.w[v] = *(const unsigned int*)&Pl[wv][l16 * 32 + kk];
        }

        // ---- O += P * V  (4 output column tiles of 16) ----
#pragma unroll
        for (int t = 0; t < 4; ++t) {
            FragBF bv;
#pragma unroll
            for (int v = 0; v < 8; ++v) {
                bv.w[v] = *(const unsigned int*)(
                    vp + (size_t)(t * 16 + l16) * 2048 + kv0 + (half << 4) + (v << 1));
            }
            O[t].v = __builtin_amdgcn_wmma_f32_16x16x32_bf16(
                false, ap.v, false, bv.v, (short)0, O[t].v, false, false);
        }
    }

    // ---- normalize and write merged-head activations ----
#pragma unroll
    for (int r = 0; r < 8; ++r) {
        float inv = 1.0f / lrow[r];
        int ss = q0 + r + half * 8;
#pragma unroll
        for (int t = 0; t < 4; ++t) {
            int col = h * 64 + t * 16 + l16;
            Aout[((size_t)(b * 2048 + ss)) * 1024 + col] = f2bf(O[t].f[r] * inv);
        }
    }
}

// ---------------------------------------------------------------------------
// Launch
// ---------------------------------------------------------------------------
extern "C" void kernel_launch(void* const* d_in, const int* in_sizes, int n_in,
                              void* d_out, int out_size, void* d_ws, size_t ws_size,
                              hipStream_t stream) {
    const float* x      = (const float*)d_in[0];
    const float* w_attn = (const float*)d_in[1];
    const float* b_attn = (const float*)d_in[2];
    const float* w_proj = (const float*)d_in[3];
    const float* b_proj = (const float*)d_in[4];
    // d_in[5]=n_state(1024), d_in[6]=n_head(16): hardcoded below.

    const int B = 2, S = 2048, D = 1024;
    const int M = B * S;               // 4096
    const size_t N_X  = (size_t)M * D;       // 4,194,304
    const size_t N_WA = (size_t)D * 3 * D;   // 3,145,728
    const size_t N_WP = (size_t)D * D;       // 1,048,576

    char* ws = (char*)d_ws;
    unsigned short* q_bf  = (unsigned short*)(ws + 0);
    unsigned short* k_bf  = (unsigned short*)(ws + (size_t)8  * 1024 * 1024);
    unsigned short* vT_bf = (unsigned short*)(ws + (size_t)16 * 1024 * 1024);
    unsigned short* x_bf  = (unsigned short*)(ws + (size_t)24 * 1024 * 1024);
    unsigned short* wa_bf = (unsigned short*)(ws + (size_t)32 * 1024 * 1024);
    unsigned short* wp_bf = (unsigned short*)(ws + (size_t)38 * 1024 * 1024);
    unsigned short* a_bf  = (unsigned short*)(ws + (size_t)40 * 1024 * 1024);

    float* out_a    = (float*)d_out;
    float* present  = out_a + N_X;  // [B,2,H,S,64] f32

    // 1) convert inputs to bf16
    cvt_f32_bf16<<<(int)((N_X  + 255) / 256), 256, 0, stream>>>(x, x_bf, (int)N_X);
    cvt_f32_bf16<<<(int)((N_WA + 255) / 256), 256, 0, stream>>>(w_attn, wa_bf, (int)N_WA);
    cvt_f32_bf16<<<(int)((N_WP + 255) / 256), 256, 0, stream>>>(w_proj, wp_bf, (int)N_WP);

    // 2) QKV projection (writes q/k/vT scratch + present f32)
    {
        dim3 grid(M / 128, (3 * D) / 64);
        gemm_bf16_kernel<<<grid, 256, 0, stream>>>(
            x_bf, wa_bf, b_attn, M, 3 * D, D, /*mode=*/0,
            q_bf, k_bf, vT_bf, present, nullptr);
    }

    // 3) flash attention -> merged-head bf16 activations
    {
        int total_waves = B * 16 * (S / 16);  // 4096
        attn_kernel<<<total_waves / 8, 256, 0, stream>>>(q_bf, k_bf, vT_bf, a_bf);
    }

    // 4) output projection -> d_out[0 .. B*S*D)
    {
        dim3 grid(M / 128, D / 64);
        gemm_bf16_kernel<<<grid, 256, 0, stream>>>(
            a_bf, wp_bf, b_proj, M, D, D, /*mode=*/1,
            q_bf, k_bf, vT_bf, present, out_a);
    }
}
